// DGCNN_77713138253982
// MI455X (gfx1250) — matrix operand order repaired and checked by reading, not compiled, not run on hardware
//
#include <hip/hip_runtime.h>
#include <hip/hip_bf16.h>

// ---------------------------------------------------------------------------
// DGCNN forward for MI455X (gfx1250): wave32 + v_wmma_f32_16x16x32_f16
// + GLOBAL_LOAD_ASYNC_TO_LDS staging for the kNN candidate tiles.
// ---------------------------------------------------------------------------

typedef _Float16 h16;
typedef __attribute__((ext_vector_type(8)))  _Float16 v8h;
typedef __attribute__((ext_vector_type(16))) _Float16 v16h;
typedef __attribute__((ext_vector_type(8)))  float    v8f;

#define NBATCH 32
#define NPTS   1024
#define KNN    20

__device__ inline v8f vzero8() {
  v8f z;
#pragma unroll
  for (int i = 0; i < 8; ++i) z[i] = 0.0f;
  return z;
}

// A fragment: 16x32 f16, row-major source with leading dim lda (halves).
// ISA layout: lanes 0-15 row M=lane, K = {0..7, 16..23}+32*kstep;
//             lanes 16-31 same rows, K = {8..15, 24..31}+32*kstep.
__device__ inline v16h load_frag_a(const h16* __restrict__ base, int lda,
                                   int mbase, int kstep, int lane) {
  const int row = mbase + (lane & 15);
  const int kb  = kstep * 32 + ((lane >> 4) << 3);
  const h16* p = base + row * lda + kb;
  v8h lo = *(const v8h*)p;
  v8h hi = *(const v8h*)(p + 16);
  v16h a;
#pragma unroll
  for (int i = 0; i < 8; ++i) { a[i] = lo[i]; a[i + 8] = hi[i]; }
  return a;
}

// B fragment: 32x16 f16 from transposed weights WT[N][Kcp].
// ISA layout: lanes 0-15 col N=lane, K = {0..7, 16..23}+32*kstep; lanes 16-31 K = {8..15, 24..31}.
__device__ inline v16h load_frag_b(const h16* __restrict__ wt, int kcp,
                                   int nbase, int kstep, int lane) {
  const int col = nbase + (lane & 15);
  const int kb  = kstep * 32 + ((lane >> 4) << 3);
  const h16* p = wt + col * kcp + kb;
  v8h lo = *(const v8h*)p;
  v8h hi = *(const v8h*)(p + 16);
  v16h b;
#pragma unroll
  for (int i = 0; i < 8; ++i) { b[i] = lo[i]; b[i + 8] = hi[i]; }
  return b;
}

__device__ inline void atomicMaxFloat(float* addr, float v) {
  unsigned int* ua = (unsigned int*)addr;
  unsigned int old = __hip_atomic_load(ua, __ATOMIC_RELAXED, __HIP_MEMORY_SCOPE_AGENT);
  while (__uint_as_float(old) < v) {
    unsigned int assumed = old;
    old = atomicCAS(ua, assumed, __float_as_uint(v));
    if (old == assumed) break;
  }
}

// ---------------------------------------------------------------------------
// Weight conversion: W fp32 [Kc][Cout] -> WT f16 [Cout][Kcp] (zero-padded K)
// ---------------------------------------------------------------------------
__global__ void convert_wt_kernel(const float* __restrict__ W, h16* __restrict__ WT,
                                  int Kc, int Cout, int Kcp) {
  int e = blockIdx.x * 256 + threadIdx.x;
  if (e >= Cout * Kcp) return;
  int c = e / Kcp, k = e - c * Kcp;
  WT[e] = (h16)(k < Kc ? W[k * Cout + c] : 0.0f);
}

// ---------------------------------------------------------------------------
// kNN (top-20 smallest squared distance, self excluded), per batch cloud.
// 256 threads = 256 query points; candidates tiled through LDS; per-thread
// top-K maintained in LDS (insertion sort). C=64 tiles are staged with
// gfx1250 GLOBAL_LOAD_ASYNC_TO_LDS_B128 (ASYNCcnt), C=3 scalar copy.
// ---------------------------------------------------------------------------
template <int C>
__global__ __launch_bounds__(256) void knn_kernel(const float* __restrict__ feat,
                                                  int fs, int* __restrict__ idxout) {
  __shared__ __align__(16) float sF[64 * C];
  __shared__ float sN[64];
  __shared__ float sD[256 * KNN];
  __shared__ int   sI[256 * KNN];

  const int tid = threadIdx.x;
  const int bb  = blockIdx.x >> 2;          // 4 blocks per cloud
  const int p   = ((blockIdx.x & 3) << 8) + tid;
  const long gp = (long)bb * NPTS + p;

  float qf[C];
  float qn = 0.0f;
#pragma unroll
  for (int c = 0; c < C; ++c) { qf[c] = feat[gp * fs + c]; qn += qf[c] * qf[c]; }

  float* md = &sD[tid * KNN];
  int*   mi = &sI[tid * KNN];
#pragma unroll
  for (int k = 0; k < KNN; ++k) { md[k] = 3.4e38f; mi[k] = 0; }

  const unsigned sFbase = (unsigned)(size_t)(void*)sF;  // LDS byte offset

  for (int t = 0; t < NPTS / 64; ++t) {
    const int cb = t * 64;
    if constexpr (C == 64) {
      // async copy: 64 rows x 64 floats = 1024 x 16B chunks, 4 per thread
#pragma unroll
      for (int pass = 0; pass < 4; ++pass) {
        const int ci  = tid + pass * 256;        // 16B chunk index
        const int row = ci >> 4;
        const int c4  = (ci & 15) << 2;          // float offset in row
        const float* g = &feat[((long)bb * NPTS + cb + row) * fs + c4];
        const unsigned l = sFbase + (unsigned)((row * 64 + c4) * 4);
        asm volatile("global_load_async_to_lds_b128 %0, %1, off"
                     :: "v"(l), "v"(g) : "memory");
      }
      asm volatile("s_wait_asynccnt 0x0" ::: "memory");
    } else {
      for (int e = tid; e < 64 * C; e += 256)
        sF[e] = feat[((long)bb * NPTS + cb + e / C) * fs + (e % C)];
    }
    // prefetch next tile while this one is consumed
    if (t + 1 < NPTS / 64)
      __builtin_prefetch(&feat[((long)bb * NPTS + cb + 64 + (tid & 63)) * fs], 0, 1);
    __syncthreads();
    if (tid < 64) {
      float s = 0.0f;
#pragma unroll
      for (int c = 0; c < C; ++c) { float v = sF[tid * C + c]; s += v * v; }
      sN[tid] = s;
    }
    __syncthreads();
    for (int q = 0; q < 64; ++q) {
      const int qg = cb + q;
      if (qg == p) continue;
      float dot = 0.0f;
#pragma unroll
      for (int c = 0; c < C; ++c) dot += qf[c] * sF[q * C + c];
      const float d = qn + sN[q] - 2.0f * dot;
      if (d < md[KNN - 1]) {
        int pos = KNN - 1;
        while (pos > 0 && md[pos - 1] > d) {
          md[pos] = md[pos - 1]; mi[pos] = mi[pos - 1]; --pos;
        }
        md[pos] = d; mi[pos] = qg;
      }
    }
    __syncthreads();
  }
#pragma unroll
  for (int k = 0; k < KNN; ++k) idxout[gp * KNN + k] = mi[k];
}

// ---------------------------------------------------------------------------
// Fused EdgeConv: build msg=[x_i, x_j-x_i] tile in LDS (f16), run 1- or
// 2-layer MLP with WMMA, bias+ReLU, max over K neighbors, write cat slice.
// Block = 4 points -> 80 message rows -> 5 M-tiles. Waves own a fixed output
// column group (nt = wave % NT) so weight B-fragments stay in registers
// across all m-tiles; the inner loop is ds_load(A) -> v_wmma only.
// ---------------------------------------------------------------------------
template <int CIN, int KCP1, int C1, int C2, bool TWO>
__global__ __launch_bounds__(256) void edgeconv_kernel(
    const float* __restrict__ feat, int fs, const int* __restrict__ idx,
    const h16* __restrict__ WT1, const float* __restrict__ b1,
    const h16* __restrict__ WT2, const float* __restrict__ b2,
    float* __restrict__ out, int os) {
  constexpr int TP = 4;
  constexpr int MR = TP * KNN;   // 80 rows
  constexpr int MT = MR / 16;    // 5 m-tiles
  __shared__ __align__(16) h16 sMsg[MR * KCP1];
  __shared__ __align__(16) h16 sY1[MR * C1];
  __shared__ __align__(16) h16 sY2[TWO ? MR * C2 : 16];

  const int tid = threadIdx.x;
  const int bb  = blockIdx.x >> 8;                 // 256 blocks per cloud
  const int p0  = (blockIdx.x & 255) * TP;
  const long gp0 = (long)bb * NPTS + p0;

  // ---- stage edge messages into LDS (f16, K-padded with zeros) ----
  for (int e = tid; e < MR * KCP1; e += 256) {
    const int r  = e / KCP1, c = e - r * KCP1;
    const int pi = r / KNN,  kk = r - pi * KNN;
    const long gi = gp0 + pi;
    float v = 0.0f;
    if (c < CIN) {
      v = feat[gi * fs + c];
    } else if (c < 2 * CIN) {
      const int  jl = idx[gi * KNN + kk];
      const long gj = (long)bb * NPTS + jl;
      const int  cc = c - CIN;
      v = feat[gj * fs + cc] - feat[gi * fs + cc];
    }
    sMsg[e] = (h16)v;
  }
  __syncthreads();

  const int wave = tid >> 5, lane = tid & 31;

  // ---- layer 1: [MR,KCP1] @ [KCP1,C1], bias + ReLU -> sY1 ----
  {
    constexpr int NT = C1 / 16;      // 4 or 8 (divides 8 waves)
    constexpr int KS = KCP1 / 32;
    constexpr int MSTEP = 8 / NT;
    const int nt  = wave % NT;
    const int col = nt * 16 + (lane & 15);
    v16h bf[KS];
#pragma unroll
    for (int ks = 0; ks < KS; ++ks) bf[ks] = load_frag_b(WT1, KCP1, nt * 16, ks, lane);
    const float bv = b1[col];
    for (int mt = wave / NT; mt < MT; mt += MSTEP) {
      v8f acc = vzero8();
#pragma unroll
      for (int ks = 0; ks < KS; ++ks) {
        v16h a = load_frag_a(sMsg, KCP1, mt * 16, ks, lane);
        acc = __builtin_amdgcn_wmma_f32_16x16x32_f16(false, a, false, bf[ks],
                                                     (short)0, acc, false, false);
      }
#pragma unroll
      for (int r = 0; r < 8; ++r) {
        const int row = mt * 16 + r + ((lane >> 4) << 3);
        sY1[row * C1 + col] = (h16)fmaxf(acc[r] + bv, 0.0f);
      }
    }
  }
  __syncthreads();

  // ---- optional layer 2: [MR,C1] @ [C1,C2], bias + ReLU -> sY2 ----
  if constexpr (TWO) {
    constexpr int NT2 = C2 / 16;
    constexpr int KS2 = C1 / 32;
    constexpr int MSTEP2 = 8 / NT2;
    const int nt  = wave % NT2;
    const int col = nt * 16 + (lane & 15);
    v16h bf[KS2];
#pragma unroll
    for (int ks = 0; ks < KS2; ++ks) bf[ks] = load_frag_b(WT2, C1, nt * 16, ks, lane);
    const float bv = b2[col];
    for (int mt = wave / NT2; mt < MT; mt += MSTEP2) {
      v8f acc = vzero8();
#pragma unroll
      for (int ks = 0; ks < KS2; ++ks) {
        v16h a = load_frag_a(sY1, C1, mt * 16, ks, lane);
        acc = __builtin_amdgcn_wmma_f32_16x16x32_f16(false, a, false, bf[ks],
                                                     (short)0, acc, false, false);
      }
#pragma unroll
      for (int r = 0; r < 8; ++r) {
        const int row = mt * 16 + r + ((lane >> 4) << 3);
        sY2[row * C2 + col] = (h16)fmaxf(acc[r] + bv, 0.0f);
      }
    }
    __syncthreads();
  }

  // ---- max over K neighbors -> out slice (ReLU output => max >= 0) ----
  constexpr int FC = TWO ? C2 : C1;
  const h16* fin = TWO ? sY2 : sY1;
  for (int e = tid; e < TP * FC; e += 256) {
    const int pi = e / FC, c = e - pi * FC;
    float m = 0.0f;
#pragma unroll 4
    for (int kk = 0; kk < KNN; ++kk)
      m = fmaxf(m, (float)fin[(pi * KNN + kk) * FC + c]);
    out[(gp0 + pi) * os + c] = m;
  }
}

// ---------------------------------------------------------------------------
// pooled init
// ---------------------------------------------------------------------------
__global__ void init_neg_kernel(float* __restrict__ p, int n) {
  int e = blockIdx.x * 256 + threadIdx.x;
  if (e < n) p[e] = -3.4e38f;
}

// ---------------------------------------------------------------------------
// g = cat @ Wa + ba (WMMA), fused with global max-pool.
// Block = 64 rows (4 m-tiles, one batch) x all 1024 cols. Each wave owns a
// fixed column group per iteration: B fragments (8 x v16h) stay in registers
// across the 4 m-tiles, so the inner loop is ds_load(A) -> v_wmma, and global
// weight traffic/waits amortize 4x. Column max accumulated over m-tiles,
// one float atomic-max per (wave, n-tile).
// ---------------------------------------------------------------------------
__global__ __launch_bounds__(256) void linear_pool_kernel(
    const float* __restrict__ cat, const h16* __restrict__ WTa,
    const float* __restrict__ ba, float* __restrict__ pooled) {
  __shared__ __align__(16) h16 sA[64 * 256];     // 32 KB
  const int tid = threadIdx.x;
  const long r0 = (long)blockIdx.x * 64;
  const int bb  = (int)(r0 / NPTS);

  for (int e = tid; e < 64 * 256; e += 256)
    sA[e] = (h16)cat[(r0 + (e >> 8)) * 256 + (e & 255)];
  __syncthreads();

  const int wave = tid >> 5, lane = tid & 31;

  for (int nt = wave; nt < 64; nt += 8) {
    // weight fragments register-resident for all 4 m-tiles
    v16h bf[8];
#pragma unroll
    for (int ks = 0; ks < 8; ++ks) bf[ks] = load_frag_b(WTa, 256, nt * 16, ks, lane);
    // prefetch next n-tile's weights while this one computes
    if (nt + 8 < 64)
      __builtin_prefetch(WTa + (nt + 8) * 16 * 256 + (lane << 4), 0, 1);
    const int col  = nt * 16 + (lane & 15);
    const float bv = ba[col];
    float m = -3.4e38f;
#pragma unroll
    for (int mt = 0; mt < 4; ++mt) {
      v8f acc = vzero8();
#pragma unroll
      for (int ks = 0; ks < 8; ++ks) {
        v16h a = load_frag_a(sA, 256, mt * 16, ks, lane);
        acc = __builtin_amdgcn_wmma_f32_16x16x32_f16(false, a, false, bf[ks],
                                                     (short)0, acc, false, false);
      }
#pragma unroll
      for (int r = 0; r < 8; ++r) m = fmaxf(m, acc[r] + bv);
    }
    m = fmaxf(m, __shfl_xor(m, 16, 32));   // combine row halves
    if (lane < 16) atomicMaxFloat(&pooled[bb * 1024 + col], m);
  }
}

// ---------------------------------------------------------------------------
// Head MLP (M=32, tiny): scalar fp32.
// ---------------------------------------------------------------------------
__global__ void head_linear_kernel(const float* __restrict__ in,
                                   const float* __restrict__ W,
                                   const float* __restrict__ bias,
                                   float* __restrict__ out,
                                   int M, int Kc, int Nc, int relu) {
  int e = blockIdx.x * 256 + threadIdx.x;
  if (e >= M * Nc) return;
  const int row = e / Nc, col = e - row * Nc;
  float s = bias[col];
  for (int k = 0; k < Kc; ++k) s += in[row * Kc + k] * W[k * Nc + col];
  if (relu) s = fmaxf(s, 0.0f);
  out[e] = s;
}

// ---------------------------------------------------------------------------
// Host launcher
// ---------------------------------------------------------------------------
extern "C" void kernel_launch(void* const* d_in, const int* in_sizes, int n_in,
                              void* d_out, int out_size, void* d_ws, size_t ws_size,
                              hipStream_t stream) {
  (void)in_sizes; (void)n_in; (void)out_size; (void)ws_size;
  const int N = NBATCH * NPTS;

  const float* x   = (const float*)d_in[0];
  const float* W0a = (const float*)d_in[2];  const float* b0a = (const float*)d_in[3];
  const float* W0b = (const float*)d_in[4];  const float* b0b = (const float*)d_in[5];
  const float* W1  = (const float*)d_in[6];  const float* b1  = (const float*)d_in[7];
  const float* W2  = (const float*)d_in[8];  const float* b2  = (const float*)d_in[9];
  const float* Wa  = (const float*)d_in[10]; const float* ba  = (const float*)d_in[11];
  const float* Wh0 = (const float*)d_in[12]; const float* bh0 = (const float*)d_in[13];
  const float* Wh1 = (const float*)d_in[14]; const float* bh1 = (const float*)d_in[15];
  const float* Wh2 = (const float*)d_in[16]; const float* bh2 = (const float*)d_in[17];

  // workspace carve-up (all offsets multiples of 256 bytes)
  char* ws = (char*)d_ws;
  size_t off = 0;
  float* cat   = (float*)(ws + off); off += (size_t)N * 256 * 4;   // [N,256]
  int*   idx   = (int*)  (ws + off); off += (size_t)N * KNN * 4;   // reused
  h16* WT0a = (h16*)(ws + off); off += (size_t)64 * 32 * 2;
  h16* WT0b = (h16*)(ws + off); off += (size_t)64 * 64 * 2;
  h16* WT1  = (h16*)(ws + off); off += (size_t)64 * 128 * 2;
  h16* WT2  = (h16*)(ws + off); off += (size_t)128 * 128 * 2;
  h16* WTa  = (h16*)(ws + off); off += (size_t)1024 * 256 * 2;
  float* pooled = (float*)(ws + off); off += (size_t)NBATCH * 1024 * 4;
  float* hh1    = (float*)(ws + off); off += (size_t)NBATCH * 512 * 4;
  float* hh2    = (float*)(ws + off); off += (size_t)NBATCH * 256 * 4;

  // 1) weight conversion to f16 transposed/padded
  convert_wt_kernel<<<(64 * 32 + 255) / 256, 256, 0, stream>>>(W0a, WT0a, 6, 64, 32);
  convert_wt_kernel<<<(64 * 64 + 255) / 256, 256, 0, stream>>>(W0b, WT0b, 64, 64, 64);
  convert_wt_kernel<<<(64 * 128 + 255) / 256, 256, 0, stream>>>(W1, WT1, 128, 64, 128);
  convert_wt_kernel<<<(128 * 128 + 255) / 256, 256, 0, stream>>>(W2, WT2, 128, 128, 128);
  convert_wt_kernel<<<(1024 * 256 + 255) / 256, 256, 0, stream>>>(Wa, WTa, 256, 1024, 256);

  const int knnGrid  = NBATCH * (NPTS / 256);
  const int convGrid = NBATCH * (NPTS / 4);

  // 2) EdgeConv 0: kNN on xyz, MLP 6->64->64, max-k  -> cat[:,0:64]
  knn_kernel<3><<<knnGrid, 256, 0, stream>>>(x, 3, idx);
  edgeconv_kernel<3, 32, 64, 64, true><<<convGrid, 256, 0, stream>>>(
      x, 3, idx, WT0a, b0a, WT0b, b0b, cat + 0, 256);

  // 3) EdgeConv 1: kNN on h0, MLP 128->64 -> cat[:,64:128]
  knn_kernel<64><<<knnGrid, 256, 0, stream>>>(cat + 0, 256, idx);
  edgeconv_kernel<64, 128, 64, 64, false><<<convGrid, 256, 0, stream>>>(
      cat + 0, 256, idx, WT1, b1, nullptr, nullptr, cat + 64, 256);

  // 4) EdgeConv 2: kNN on h1, MLP 128->128 -> cat[:,128:256]
  knn_kernel<64><<<knnGrid, 256, 0, stream>>>(cat + 64, 256, idx);
  edgeconv_kernel<64, 128, 128, 64, false><<<convGrid, 256, 0, stream>>>(
      cat + 64, 256, idx, WT2, b2, nullptr, nullptr, cat + 128, 256);

  // 5) g = cat @ Wa + ba, max-pool over points
  init_neg_kernel<<<(NBATCH * 1024 + 255) / 256, 256, 0, stream>>>(pooled, NBATCH * 1024);
  linear_pool_kernel<<<N / 64, 256, 0, stream>>>(cat, WTa, ba, pooled);

  // 6) head MLP
  head_linear_kernel<<<(NBATCH * 512 + 255) / 256, 256, 0, stream>>>(
      pooled, Wh0, bh0, hh1, NBATCH, 1024, 512, 1);
  head_linear_kernel<<<(NBATCH * 256 + 255) / 256, 256, 0, stream>>>(
      hh1, Wh1, bh1, hh2, NBATCH, 512, 256, 1);
  head_linear_kernel<<<(NBATCH * 40 + 255) / 256, 256, 0, stream>>>(
      hh2, Wh2, bh2, (float*)d_out, NBATCH, 256, 40, 0);
}
